// LoRAAdapter_67688684585121
// MI455X (gfx1250) — matrix operand order repaired and checked by reading, not compiled, not run on hardware
//
#include <hip/hip_runtime.h>
#include <hip/hip_bf16.h>
#include <math.h>

// Problem constants (match reference)
#define BB   4
#define SS   2048
#define DD   4096
#define OUT  4096
#define EE   8
#define RR   32
// SCALING = 2.0 folded into Wc

typedef __attribute__((ext_vector_type(16))) __bf16 v16bf;
typedef __attribute__((ext_vector_type(8)))  __bf16 v8bf;
typedef __attribute__((ext_vector_type(8)))  float  v8f;

// ---------------- workspace layout ----------------
// [0,128)            gates  float[B*E]
// [256, +256KB)      Awb    bf16[R*D]
// [262400, +1MB)     Wc     bf16[B*OUT*R]   (pre-scaled by 2.0)
// [1310976, +512KB)  Sh     bf16[B*S*R]     (shared low-rank activations)
#define WS_AWB_OFF    256
#define WS_WC_OFF     262400
#define WS_SH_OFF     1310976

// ============================================================
// Kernel 1: noisy top-2 gating.  One block of 256 threads.
// ============================================================
__global__ __launch_bounds__(256) void gating_kernel(
    const float* __restrict__ x, const int* __restrict__ eof,
    const float* __restrict__ noise, const float* __restrict__ route_w,
    const float* __restrict__ noise_w, float* __restrict__ gates_out)
{
    __shared__ float sLogit[BB][EE];
    __shared__ float sGate[BB][EE];
    const int tid  = threadIdx.x;
    const int wave = tid >> 5;
    const int lane = tid & 31;

    // 32 (b,e) pairs over 8 waves, 4 pairs per wave
    for (int p = wave; p < BB * EE; p += 8) {
        const int b = p >> 3, e = p & 7;
        const float* xr = x + ((size_t)b * SS + (size_t)eof[b]) * DD;
        const float* rw = route_w + (size_t)e * DD;
        const float* nw = noise_w + (size_t)e * DD;
        float ac = 0.f, an = 0.f;
        for (int d = lane; d < DD; d += 32) {
            float xv = xr[d];
            ac += xv * rw[d];
            an += xv * nw[d];
        }
        #pragma unroll
        for (int off = 16; off > 0; off >>= 1) {
            ac += __shfl_xor(ac, off);
            an += __shfl_xor(an, off);
        }
        if (lane == 0) {
            float sp = (an > 20.f) ? an : log1pf(expf(an));   // softplus
            sLogit[b][e] = ac + noise[b * EE + e] * (sp + 0.01f);
        }
    }
    if (tid < BB * EE) sGate[tid >> 3][tid & 7] = 0.f;
    __syncthreads();

    if (tid < BB) {
        const int b = tid;
        float lg[EE];
        #pragma unroll
        for (int e = 0; e < EE; ++e) lg[e] = sLogit[b][e];
        int i1 = 0;
        #pragma unroll
        for (int e = 1; e < EE; ++e) if (lg[e] > lg[i1]) i1 = e;
        int i2 = (i1 == 0) ? 1 : 0;
        #pragma unroll
        for (int e = 0; e < EE; ++e) if (e != i1 && lg[e] > lg[i2]) i2 = e;
        // softmax over top-2 (i1 is the max)
        float e2  = expf(lg[i2] - lg[i1]);
        float inv = 1.f / (1.f + e2);
        sGate[b][i1] = inv;
        sGate[b][i2] = e2 * inv;
    }
    __syncthreads();
    if (tid < BB * EE) gates_out[tid] = sGate[tid >> 3][tid & 7];
}

// ============================================================
// Kernel 2: A_w (f32) -> bf16
// ============================================================
__global__ __launch_bounds__(256) void cvt_aw_kernel(
    const float* __restrict__ A_w, __bf16* __restrict__ Awb)
{
    int idx = blockIdx.x * 256 + threadIdx.x;   // < R*D = 131072
    Awb[idx] = (__bf16)A_w[idx];
}

// ============================================================
// Kernel 3: Wc[b,o,r] = 2.0 * sum_e gates[b,e] * B_w[e,o,r]  (bf16)
// ============================================================
__global__ __launch_bounds__(256) void build_wc_kernel(
    const float* __restrict__ B_w, const float* __restrict__ gates,
    __bf16* __restrict__ Wc)
{
    int idx = blockIdx.x * 256 + threadIdx.x;   // < B*OUT*R = 524288
    int b   = idx >> 17;                        // OUT*R = 131072 = 2^17
    int rem = idx & (131072 - 1);
    float acc = 0.f;
    #pragma unroll
    for (int e = 0; e < EE; ++e)
        acc += gates[b * EE + e] * B_w[(size_t)e * 131072 + rem];
    Wc[idx] = (__bf16)(acc * 2.0f);             // fold SCALING here
}

// ============================================================
// Kernel 4: shared = x @ A_w^T  (split-K over 8 waves, WMMA bf16)
// One block per 16-row s-tile.  Output bf16 tile [16][32] to ws.
// ============================================================
__global__ __launch_bounds__(256) void shared_proj_kernel(
    const float* __restrict__ x, const __bf16* __restrict__ Awb,
    __bf16* __restrict__ Sh)
{
    __shared__ float part[8][16][RR];           // 16 KB split-K partials
    const int tid  = threadIdx.x;
    const int wave = tid >> 5;
    const int lane = tid & 31;
    const int hi   = lane >> 4;                 // half-wave select
    const int nl   = lane & 15;                 // M (A) / N (B) index
    const size_t tile = blockIdx.x;             // 16-row tile of flattened [B*S]

    const float* xrow = x + (tile * 16 + (size_t)nl) * DD;

    v8f clo = {};                               // C for r = 0..15
    v8f chi = {};                               // C for r = 16..31
    const int k0base = wave * (DD / 8);         // 512 K per wave

    for (int ks = 0; ks < 16; ++ks) {
        const int k0 = k0base + ks * 32;
        // pull the next iteration's x cache line toward the WGP early
        if (ks < 15) __builtin_prefetch(xrow + k0 + 32 + hi * 16, 0, 0);
        // ---- A fragment (x, f32 -> bf16), 16x32 layout:
        // VGPR j holds K pair base = 2j + hi*16 (contiguous)
        const float* p = xrow + k0 + hi * 16;
        float4 f0 = *(const float4*)(p);
        float4 f1 = *(const float4*)(p + 4);
        float4 f2 = *(const float4*)(p + 8);
        float4 f3 = *(const float4*)(p + 12);
        v16bf a;
        a[0]=(__bf16)f0.x; a[1]=(__bf16)f0.y; a[2]=(__bf16)f0.z; a[3]=(__bf16)f0.w;
        a[4]=(__bf16)f1.x; a[5]=(__bf16)f1.y; a[6]=(__bf16)f1.z; a[7]=(__bf16)f1.w;
        a[8]=(__bf16)f2.x; a[9]=(__bf16)f2.y; a[10]=(__bf16)f2.z; a[11]=(__bf16)f2.w;
        a[12]=(__bf16)f3.x; a[13]=(__bf16)f3.y; a[14]=(__bf16)f3.z; a[15]=(__bf16)f3.w;
        // ---- B fragments (A_w bf16), 32x16 layout:
        // lanes 0-15 hold K=k0..k0+15, lanes 16-31 hold K=k0+16..k0+31 (contiguous)
        v16bf b0 = *(const v16bf*)(Awb + (size_t)nl * DD + k0 + hi * 16);
        v16bf b1 = *(const v16bf*)(Awb + (size_t)(nl + 16) * DD + k0 + hi * 16);
        clo = __builtin_amdgcn_wmma_f32_16x16x32_bf16(false, a, false, b0,
                                                      (short)0, clo, false, false);
        chi = __builtin_amdgcn_wmma_f32_16x16x32_bf16(false, a, false, b1,
                                                      (short)0, chi, false, false);
    }
    // C/D layout: element v <-> (M = v + hi*8, N = nl)
    #pragma unroll
    for (int v = 0; v < 8; ++v) {
        const int m = v + hi * 8;
        part[wave][m][nl]      = clo[v];
        part[wave][m][nl + 16] = chi[v];
    }
    __syncthreads();
    // reduce split-K partials: 512 values, 256 threads
    for (int i = tid; i < 16 * RR; i += 256) {
        const int m = i >> 5, r = i & 31;
        float s = 0.f;
        #pragma unroll
        for (int w = 0; w < 8; ++w) s += part[w][m][r];
        Sh[(tile * 16 + m) * RR + r] = (__bf16)s;
    }
}

// ============================================================
// Kernel 5: out = Sh @ Wc[b]^T   (one WMMA per 16x16 out tile, K=R=32)
// grid = (B*S/16) * (OUT/256); 8 waves * 2 o-tiles each.
// Output staged through LDS for fully coalesced b128 stores.
// ============================================================
#define OPAD 4   // dword padding per row to avoid half-wave bank conflicts
__global__ __launch_bounds__(256) void expand_kernel(
    const __bf16* __restrict__ Sh, const __bf16* __restrict__ Wc,
    float* __restrict__ out)
{
    __shared__ float otile[16][256 + OPAD];     // ~16.3 KB output staging
    const int tid  = threadIdx.x;
    const int wave = tid >> 5;
    const int lane = tid & 31;
    const int hi   = lane >> 4;
    const int nl   = lane & 15;
    const int tile = blockIdx.x >> 4;           // 16-row s-tile
    const int oblk = blockIdx.x & 15;           // 256-wide OUT block
    const int b    = (tile * 16) / SS;
    const int o0   = oblk * 256 + wave * 32;

    // A fragment from shared bf16 tile [16][32], row = nl
    const __bf16* arow = Sh + ((size_t)tile * 16 + nl) * RR;
    v8bf alo = *(const v8bf*)(arow + hi * 16);       // K pairs for VGPR 0..3
    v8bf ahi = *(const v8bf*)(arow + hi * 16 + 8);   // K pairs for VGPR 4..7
    v16bf a;
    #pragma unroll
    for (int i = 0; i < 8; ++i) { a[i] = alo[i]; a[8 + i] = ahi[i]; }

    #pragma unroll
    for (int ot = 0; ot < 2; ++ot) {
        const int o = o0 + ot * 16 + nl;
        // B fragment: Wc row o, contiguous 16 K values per half-wave
        v16bf bf = *(const v16bf*)(Wc + ((size_t)b * OUT + o) * RR + hi * 16);
        v8f c = {};
        c = __builtin_amdgcn_wmma_f32_16x16x32_bf16(false, a, false, bf,
                                                    (short)0, c, false, false);
        const int col = wave * 32 + ot * 16 + nl;   // column within block
        #pragma unroll
        for (int v = 0; v < 8; ++v) {
            otile[v + hi * 8][col] = c[v];
        }
    }
    __syncthreads();

    // coalesced writeback: 16 rows x 256 f32 = 1024 float4, 4 per thread
    float* obase = out + (size_t)tile * 16 * OUT + oblk * 256;
    #pragma unroll
    for (int i = tid; i < 1024; i += 256) {
        const int row = i >> 6;                 // 64 float4 per row
        const int c4  = i & 63;
        float4 vdata = *(const float4*)&otile[row][c4 * 4];
        *(float4*)(obase + (size_t)row * OUT + c4 * 4) = vdata;
    }
}

// ============================================================
extern "C" void kernel_launch(void* const* d_in, const int* in_sizes, int n_in,
                              void* d_out, int out_size, void* d_ws, size_t ws_size,
                              hipStream_t stream)
{
    const float* x       = (const float*)d_in[0];
    const int*   eof     = (const int*)  d_in[1];
    const float* noise   = (const float*)d_in[2];
    const float* A_w     = (const float*)d_in[3];
    const float* B_w     = (const float*)d_in[4];
    const float* route_w = (const float*)d_in[5];
    const float* noise_w = (const float*)d_in[6];
    float* out = (float*)d_out;

    char* ws = (char*)d_ws;
    float*  gates = (float*) (ws);
    __bf16* Awb   = (__bf16*)(ws + WS_AWB_OFF);
    __bf16* Wc    = (__bf16*)(ws + WS_WC_OFF);
    __bf16* Sh    = (__bf16*)(ws + WS_SH_OFF);

    gating_kernel   <<<1,    256, 0, stream>>>(x, eof, noise, route_w, noise_w, gates);
    cvt_aw_kernel   <<<512,  256, 0, stream>>>(A_w, Awb);               // R*D/256
    build_wc_kernel <<<2048, 256, 0, stream>>>(B_w, gates, Wc);         // B*OUT*R/256
    shared_proj_kernel<<<BB * SS / 16, 256, 0, stream>>>(x, Awb, Sh);   // 512 tiles
    expand_kernel   <<<(BB * SS / 16) * (OUT / 256), 256, 0, stream>>>(Sh, Wc, out);
}
